// BBBLSTM_60696477827102
// MI455X (gfx1250) — matrix operand order repaired and checked by reading, not compiled, not run on hardware
//
#include <hip/hip_runtime.h>

// ---------------------------------------------------------------------------
// Fused LayerNorm + Bayesian-LSTM (T=30) + MLP head for MI455X (gfx1250).
// 512 threads (16 wave32) per block own 16 batch rows. Each wave owns 2 of
// the 32 gate N-tiles and keeps ALL its WMMA B-fragments (w_ih K-padded with
// the bias row folded in + w_hh) in registers for the whole recurrence, so
// the 30-step loop issues v_wmma_f32_16x16x32_f16 back-to-back with only
// A-fragment ds_loads in between. Gate activations use v_exp_f32 + v_rcp_f32
// (TRANS ops) only. LDS holds just x (layernormed, f16), h, and the per-step
// gate tile: 66 KB.
// ---------------------------------------------------------------------------

typedef __attribute__((ext_vector_type(16))) _Float16 v16h;
typedef __attribute__((ext_vector_type(8)))  _Float16 v8h;
typedef __attribute__((ext_vector_type(8)))  float    v8f;
typedef __attribute__((ext_vector_type(4)))  float    v4f;

#define BDIM   8192
#define TDIM   30
#define FDIM   25
#define HDIM   128
#define GDIM   512      // 4*H
#define MTILE  16       // batch rows per block
#define KB     4        // HDIM/32
#define NTHREADS 512    // 16 wave32
#define TPW    2        // N-tiles per wave (16 waves * 2 = 32 tiles = 4H)

// LDS (half-element offsets):
//   xall : [t][m][32] layernormed x, f16, col25=1.0 (bias row), 15360 halves
//   hbuf : [m][128]   current h, f16, 2048 halves
//   gates: [nt][lane][8] f32 C-fragments / MLP scratch, 8192 floats
#define OFF_HBUF (TDIM * MTILE * 32)
#define OFF_END  (OFF_HBUF + MTILE * HDIM)
#define SMEM_BYTES (OFF_END * 2 + 8192 * 4)   // 67584 B

// fast activations: v_exp_f32 + v_rcp_f32 (single TRANS ops, no IEEE div chain)
__device__ __forceinline__ float sigf(float x) {
    return __builtin_amdgcn_rcpf(1.0f + __expf(-x));
}
__device__ __forceinline__ float tanhfast(float x) {
    return 1.0f - 2.0f * __builtin_amdgcn_rcpf(__expf(2.0f * x) + 1.0f);
}

__global__ __launch_bounds__(NTHREADS, 1)
void bbb_lstm_fused(const float* __restrict__ x,
                    const float* __restrict__ ln_g,  const float* __restrict__ ln_b,
                    const float* __restrict__ w_ih,  const float* __restrict__ w_hh,
                    const float* __restrict__ b_lstm,
                    const float* __restrict__ w1, const float* __restrict__ b1,
                    const float* __restrict__ w2, const float* __restrict__ b2,
                    const float* __restrict__ w3, const float* __restrict__ b3,
                    float* __restrict__ out)
{
    extern __shared__ char smem[];
    _Float16* xall = (_Float16*)smem;
    _Float16* hbuf = xall + OFF_HBUF;
    float*    gatesF = (float*)(xall + OFF_END);

    const int tid  = threadIdx.x;
    const int lane = tid & 31;              // wave32
    const int wv   = tid >> 5;              // 0..15
    const int b0   = blockIdx.x * MTILE;
    const int halfSel = lane >> 4;          // K-half selector (A/B f16 layouts)
    const int mrow    = lane & 15;

    // ---- loop-invariant B fragments, gathered straight from global ---------
    // B 32x16 f16 layout: lanes 0-15 hold K 0-7/16-23, lanes 16-31
    // hold K 8-15/24-31, column N = 16*nt + (lane&15).
    // w_ih is K-padded 25->32 with row 25 = b_lstm (bias folded into the GEMM).
    v16h bx[TPW];
    v16h bh[KB][TPW];
#pragma unroll
    for (int i = 0; i < TPW; ++i) {
        const int nt = TPW * wv + i;
        const int n  = 16 * nt + mrow;
#pragma unroll
        for (int j = 0; j < 16; ++j) {
            const int k = 8 * halfSel + (j & 7) + 16 * (j >> 3); // 0..31
            float v;
            if (k < FDIM)       v = w_ih[k * GDIM + n];
            else if (k == FDIM) v = b_lstm[n];
            else                v = 0.0f;
            bx[i][j] = (_Float16)v;
        }
#pragma unroll
        for (int kb = 0; kb < KB; ++kb)
#pragma unroll
            for (int j = 0; j < 16; ++j) {
                const int k = 32 * kb + 8 * halfSel + (j & 7) + 16 * (j >> 3);
                bh[kb][i][j] = (_Float16)w_hh[k * GDIM + n];
            }
    }

    // ---- all 30*16 layernorms up front (off the recurrence critical path) --
    if (tid < TDIM * MTILE) {
        const int tt = tid >> 4;            // 0..29
        const int m  = tid & 15;
        const float* xp = x + ((size_t)(b0 + m) * TDIM + tt) * FDIM;
        float s = 0.f, sq = 0.f;
#pragma unroll
        for (int f = 0; f < FDIM; ++f) { const float v = xp[f]; s += v; sq += v * v; }
        const float mu   = s * (1.0f / FDIM);
        const float var  = sq * (1.0f / FDIM) - mu * mu;
        const float rstd = rsqrtf(var + 1e-5f);
        _Float16* xr = &xall[(tt * MTILE + m) * 32];
#pragma unroll
        for (int f = 0; f < FDIM; ++f)
            xr[f] = (_Float16)((xp[f] - mu) * rstd * ln_g[f] + ln_b[f]);
        xr[FDIM] = (_Float16)1.0f;          // multiplies the folded bias row
#pragma unroll
        for (int f = FDIM + 1; f < 32; ++f) xr[f] = (_Float16)0.0f;
    }
    // h0 = 0
    for (int e = tid; e < MTILE * HDIM; e += NTHREADS) hbuf[e] = (_Float16)0.0f;
    __syncthreads();

    // per-thread cell state in C-fragment coordinates: q = wv&7 picks the
    // i/f/g/o tile quadruple (q, q+8, q+16, q+24); rbase = (wv>>3)*4 picks
    // which half of the 8 fragment rows; element r -> C slot (lane, rbase+r)
    // = matrix element (M = rbase+r+8*(lane>>4), N = 16*q + (lane&15)).
    const int q     = wv & 7;
    const int rbase = (wv >> 3) * 4;
    float c[4], hreg[4];
#pragma unroll
    for (int r = 0; r < 4; ++r) { c[r] = 0.f; hreg[r] = 0.f; }

    const v8f zero = {};

    for (int t = 0; t < TDIM; ++t) {
        __syncthreads();  // hbuf(t-1) visible; gates(t-1) fully consumed

        // -- A fragments (16-bit A 16x32 layout: lanes 0-15 K0-7/16-23,
        //    lanes 16-31 K8-15/24-31); shared by this wave's 2 N-tiles -------
        v16h ax;
        {
            const _Float16* p = &xall[(t * MTILE + mrow) * 32 + 8 * halfSel];
            const v8h lo = *(const v8h*)p;
            const v8h hi = *(const v8h*)(p + 16);
#pragma unroll
            for (int j = 0; j < 8; ++j) { ax[j] = lo[j]; ax[j + 8] = hi[j]; }
        }
        v16h ah[KB];
#pragma unroll
        for (int kb = 0; kb < KB; ++kb) {
            const _Float16* p = &hbuf[mrow * HDIM + 32 * kb + 8 * halfSel];
            const v8h lo = *(const v8h*)p;
            const v8h hi = *(const v8h*)(p + 16);
#pragma unroll
            for (int j = 0; j < 8; ++j) { ah[kb][j] = lo[j]; ah[kb][j + 8] = hi[j]; }
        }

        // -- 10 WMMAs, all operands register-resident, 2 interleaved chains --
        v8f acc[TPW];
#pragma unroll
        for (int i = 0; i < TPW; ++i)
            acc[i] = __builtin_amdgcn_wmma_f32_16x16x32_f16(
                         false, ax, false, bx[i], (short)0, zero, false, false);
#pragma unroll
        for (int kb = 0; kb < KB; ++kb)
#pragma unroll
            for (int i = 0; i < TPW; ++i)
                acc[i] = __builtin_amdgcn_wmma_f32_16x16x32_f16(
                             false, ah[kb], false, bh[kb][i], (short)0, acc[i], false, false);

        // gate tile in C-fragment order: 2x ds_store_b128 per tile
#pragma unroll
        for (int i = 0; i < TPW; ++i)
            *(v8f*)&gatesF[((TPW * wv + i) * 32 + lane) * 8] = acc[i];
        __syncthreads();  // gates(t) visible

        // -- elementwise LSTM cell update (16B vector gate reads) -------------
        {
            const v4f iV = *(const v4f*)&gatesF[((q     ) * 32 + lane) * 8 + rbase];
            const v4f fV = *(const v4f*)&gatesF[((q +  8) * 32 + lane) * 8 + rbase];
            const v4f gV = *(const v4f*)&gatesF[((q + 16) * 32 + lane) * 8 + rbase];
            const v4f oV = *(const v4f*)&gatesF[((q + 24) * 32 + lane) * 8 + rbase];
#pragma unroll
            for (int r = 0; r < 4; ++r) {
                const float cv = sigf(fV[r]) * c[r] + sigf(iV[r]) * tanhfast(gV[r]);
                c[r] = cv;
                const float h = sigf(oV[r]) * tanhfast(cv);
                hreg[r] = h;
                const int M = rbase + r + 8 * halfSel;
                const int N = 16 * q + mrow;
                hbuf[M * HDIM + N] = (_Float16)h;
            }
        }
    }

    __syncthreads();  // last gate reads done; gatesF becomes MLP scratch
    // final h (f32) row-major at gatesF[0..2047]
#pragma unroll
    for (int r = 0; r < 4; ++r)
        gatesF[(rbase + r + 8 * halfSel) * HDIM + 16 * q + mrow] = hreg[r];
    __syncthreads();

    float* y1 = gatesF + 2048;   // 16x128
    float* y2 = gatesF + 4096;   // 16x64
    const int em = tid & 15;

    // y1 = relu(h @ w1 + b1) : 16x128, 4 outputs/thread
    {
        const int jb = (tid >> 4) * 4;      // 0..124
        for (int jj = 0; jj < 4; ++jj) {
            const int j = jb + jj;
            float s = b1[j];
            for (int k = 0; k < HDIM; ++k) s += gatesF[em * HDIM + k] * w1[k * HDIM + j];
            y1[em * HDIM + j] = fmaxf(s, 0.f);
        }
    }
    __syncthreads();

    // y2 = relu(y1 @ w2 + b2) : 16x64, 2 outputs/thread
    {
        const int jb = (tid >> 4) * 2;      // 0..62
        for (int jj = 0; jj < 2; ++jj) {
            const int j = jb + jj;
            float s = b2[j];
            for (int k = 0; k < HDIM; ++k) s += y1[em * HDIM + k] * w2[k * 64 + j];
            y2[em * 64 + j] = fmaxf(s, 0.f);
        }
    }
    __syncthreads();

    // out = y2 @ w3 + b3 : 16x2
    if (tid < 32) {
        const int m = tid & 15, col = tid >> 4;
        float s = b3[col];
        for (int k = 0; k < 64; ++k) s += y2[m * 64 + k] * w3[k * 2 + col];
        out[(size_t)(b0 + m) * 2 + col] = s;
    }
}

extern "C" void kernel_launch(void* const* d_in, const int* in_sizes, int n_in,
                              void* d_out, int out_size, void* d_ws, size_t ws_size,
                              hipStream_t stream) {
    (void)in_sizes; (void)n_in; (void)out_size; (void)d_ws; (void)ws_size;
    const float* x      = (const float*)d_in[0];
    const float* ln_g   = (const float*)d_in[1];
    const float* ln_b   = (const float*)d_in[2];
    const float* w_ih   = (const float*)d_in[3];
    const float* w_hh   = (const float*)d_in[4];
    const float* b_lstm = (const float*)d_in[5];
    const float* w1     = (const float*)d_in[6];
    const float* b1     = (const float*)d_in[7];
    const float* w2     = (const float*)d_in[8];
    const float* b2     = (const float*)d_in[9];
    const float* w3     = (const float*)d_in[10];
    const float* b3     = (const float*)d_in[11];
    float* out = (float*)d_out;

    dim3 grid(BDIM / MTILE);     // 512 blocks, each owns 16 batch rows
    dim3 block(NTHREADS);        // 16 wave32
    bbb_lstm_fused<<<grid, block, SMEM_BYTES, stream>>>(
        x, ln_g, ln_b, w_ih, w_hh, b_lstm, w1, b1, w2, b2, w3, b3, out);
}
